// SelfAttention_29703993819971
// MI455X (gfx1250) — matrix operand order repaired and checked by reading, not compiled, not run on hardware
//
#include <hip/hip_runtime.h>
#include <hip/hip_bf16.h>

// ---------------------------------------------------------------------------
// SelfAttention on MI455X (gfx1250, wave32, WMMA)
//
// Algebraic reduction: scores[i,j] = mask ? s_i : -1000, so softmax row i has
// two weights (w1_i, w0_i) and out = (w1-w0) * (M@V) + w0 * colsum(V).
// Heavy path = M@V done with v_wmma_f32_16x16x32_bf16 (mask exact in bf16,
// V split into bf16 hi+lo for ~fp32 accuracy). V^T built with
// v_wmma_f32_16x16x4_f32. Mask (256 MB) streamed nontemporally once.
// ---------------------------------------------------------------------------

#define N_TOK   8192
#define DDIM    256
#define NEG_FILL (-1000.0f)
#define SLOPE    0.01f
#define NSPLIT   4
#define KSPAN    (N_TOK / NSPLIT)   // 2048

typedef __attribute__((ext_vector_type(2)))  float        v2f;
typedef __attribute__((ext_vector_type(8)))  float        v8f;
typedef __attribute__((ext_vector_type(16))) __bf16       v16bf;
typedef __attribute__((ext_vector_type(4)))  unsigned int v4u;
typedef __attribute__((ext_vector_type(8)))  unsigned int v8u;

// ---------------------------------------------------------------------------
// K1: t[e] = sum_d a[d]*Wq[d,e] + a[D+d]*Wk[d,e]   (1 block, trivial)
// ---------------------------------------------------------------------------
__global__ void k1_fold_a(const float* __restrict__ a,
                          const float* __restrict__ Wq,
                          const float* __restrict__ Wk,
                          float* __restrict__ t) {
  int e = threadIdx.x;
  float acc = 0.f;
  for (int d = 0; d < DDIM; ++d)
    acc += a[d] * Wq[d * DDIM + e] + a[DDIM + d] * Wk[d * DDIM + e];
  t[e] = acc;
}

// ---------------------------------------------------------------------------
// K2: s[i] = leakyrelu(query[i] . t)   one wave per row
// ---------------------------------------------------------------------------
__global__ void k2_scores(const float* __restrict__ query,
                          const float* __restrict__ t,
                          float* __restrict__ s) {
  __shared__ float tl[DDIM];
  tl[threadIdx.x] = t[threadIdx.x];
  __syncthreads();
  const int wave = threadIdx.x >> 5;
  const int lane = threadIdx.x & 31;
  const int row  = blockIdx.x * 8 + wave;
  const float* q = query + (long)row * DDIM;
  float acc = 0.f;
#pragma unroll
  for (int j = 0; j < 8; ++j) acc += q[lane + 32 * j] * tl[lane + 32 * j];
#pragma unroll
  for (int off = 16; off >= 1; off >>= 1) acc += __shfl_xor(acc, off, 32);
  if (lane == 0) s[row] = acc > 0.f ? acc : SLOPE * acc;
}

// ---------------------------------------------------------------------------
// K3: Vt = (query @ Wv^T)^T, stored as split bf16 (hi/lo), layout [d][i].
// Computed directly transposed: Vt[d,i] = sum_e Wv[d,e] * query[i,e]
// via v_wmma_f32_16x16x4_f32. Wave: 16 d-rows x 128 i-cols (8 C tiles).
// ---------------------------------------------------------------------------
__global__ __launch_bounds__(256, 1) void k3_vt(
    const float* __restrict__ query,
    const float* __restrict__ Wv,
    __bf16* __restrict__ vt_hi,
    __bf16* __restrict__ vt_lo) {
  const int lane = threadIdx.x & 31;
  const int w    = blockIdx.x * 8 + (threadIdx.x >> 5);  // 0..1023
  const int m0   = (w >> 6) * 16;                        // d tile base
  const int n0   = (w & 63) * 128;                       // i chunk base
  const int lm   = lane & 15;
  const int lh   = lane >> 4;                            // 0/1

  v8f acc[8] = {};
  const float* arow = Wv + (long)(m0 + lm) * DDIM + 2 * lh;

  for (int k = 0; k < DDIM; k += 4) {
    v2f a = *(const v2f*)(arow + k);
#pragma unroll
    for (int tt = 0; tt < 8; ++tt) {
      const float* bp = query + (long)(n0 + tt * 16 + lm) * DDIM + k + 2 * lh;
      v2f b = *(const v2f*)bp;
      acc[tt] = __builtin_amdgcn_wmma_f32_16x16x4_f32(
          false, a, false, b, (short)0, acc[tt], false, false);
    }
  }
  // write out split-bf16, transposed layout [d][i] (i contiguous)
#pragma unroll
  for (int tt = 0; tt < 8; ++tt) {
#pragma unroll
    for (int g = 0; g < 8; ++g) {
      float v   = acc[tt][g];
      __bf16 hi = (__bf16)v;
      __bf16 lo = (__bf16)(v - (float)hi);
      long idx  = (long)(m0 + g + 8 * lh) * N_TOK + (n0 + tt * 16 + lm);
      vt_hi[idx] = hi;
      vt_lo[idx] = lo;
    }
  }
}

// ---------------------------------------------------------------------------
// K4: colsum[d] = sum_i V[i,d] = sum_i (vt_hi[d,i] + vt_lo[d,i])
// one block per column d, deterministic tree reduction
// ---------------------------------------------------------------------------
__global__ void k4_colsum(const __bf16* __restrict__ vt_hi,
                          const __bf16* __restrict__ vt_lo,
                          float* __restrict__ colsum) {
  __shared__ float red[256];
  const int d   = blockIdx.x;
  const int tid = threadIdx.x;
  const __bf16* h = vt_hi + (long)d * N_TOK;
  const __bf16* l = vt_lo + (long)d * N_TOK;
  float acc = 0.f;
  for (int i = tid; i < N_TOK; i += 256) acc += (float)h[i] + (float)l[i];
  red[tid] = acc;
  __syncthreads();
  for (int off = 128; off >= 1; off >>= 1) {
    if (tid < off) red[tid] += red[tid + off];
    __syncthreads();
  }
  if (tid == 0) colsum[d] = red[0];
}

// ---------------------------------------------------------------------------
// K5: P_sp = M[:, ksp] @ V[ksp, :]  and  c_sp = rowsum(M[:, ksp])
// Wave: 32 rows x 256 cols, K span 2048. Mask loaded nontemporal straight
// into WMMA A layout (each element read exactly once from HBM); V^T hi/lo
// read from L2 in B layout. acc = wmma(A,Bhi,acc); acc = wmma(A,Blo,acc).
// ---------------------------------------------------------------------------
__global__ __launch_bounds__(256, 1) void k5_maskgemm(
    const int* __restrict__ mask,
    const __bf16* __restrict__ vt_hi,
    const __bf16* __restrict__ vt_lo,
    float* __restrict__ Pbuf,   // [NSPLIT][N_TOK][DDIM]
    int*   __restrict__ cbuf) { // [NSPLIT][N_TOK]
  const int lane = threadIdx.x & 31;
  const int w    = blockIdx.x * 8 + (threadIdx.x >> 5);  // 0..1023
  const int rg   = w & 255;                              // row group (32 rows)
  const int sp   = w >> 8;                               // k-split 0..3
  const int r0   = rg * 32;
  const int k0   = sp * KSPAN;
  const int lm   = lane & 15;
  const int lh   = lane >> 4;

  v8f acc[2][16] = {};
  int crow[2] = {0, 0};

  // A operand (16-bit 16x32), contiguous k order chosen here, same as B: lane<16 -> K 0..15 of row lm;
  // lane>=16 -> K 16..31. One 16-int segment per row tile.
  const unsigned int* mrow0 =
      (const unsigned int*)mask + (long)(r0 + lm) * N_TOK + k0 + lh * 16;
  const unsigned int* mrow1 = mrow0 + 16L * N_TOK;

  for (int ks = 0; ks < KSPAN; ks += 32) {
    v16bf A[2];
#pragma unroll
    for (int rt = 0; rt < 2; ++rt) {
      const unsigned int* mp = (rt ? mrow1 : mrow0) + ks;
      v4u m0a = __builtin_nontemporal_load((const v4u*)(mp));
      v4u m0b = __builtin_nontemporal_load((const v4u*)(mp + 4));
      v4u m1a = __builtin_nontemporal_load((const v4u*)(mp + 8));
      v4u m1b = __builtin_nontemporal_load((const v4u*)(mp + 12));
      unsigned int sA[16];
#pragma unroll
      for (int j = 0; j < 4; ++j) {
        sA[j] = m0a[j]; sA[4 + j] = m0b[j];
        sA[8 + j] = m1a[j]; sA[12 + j] = m1b[j];
      }
      v8u pk;
      int rs = 0;
#pragma unroll
      for (int j = 0; j < 8; ++j) {
        // 0/1 int -> packed bf16 pair: 1.0bf16 = 0x3F80
        pk[j] = sA[2 * j] * 0x3F80u + sA[2 * j + 1] * 0x3F800000u;
        rs += (int)(sA[2 * j] + sA[2 * j + 1]);
      }
      crow[rt] += rs;
      A[rt] = __builtin_bit_cast(v16bf, pk);
    }
#pragma unroll
    for (int ct = 0; ct < 16; ++ct) {
      // B operand (16-bit 32x16), contiguous k order chosen here, same as A: lane holds col n=lm, K = 16*lh + 0..15
      long boff = (long)(ct * 16 + lm) * N_TOK + k0 + ks + lh * 16;
      v16bf Bh = __builtin_bit_cast(v16bf, *(const v8u*)(vt_hi + boff));
      v16bf Bl = __builtin_bit_cast(v16bf, *(const v8u*)(vt_lo + boff));
#pragma unroll
      for (int rt = 0; rt < 2; ++rt) {
        acc[rt][ct] = __builtin_amdgcn_wmma_f32_16x16x32_bf16(
            false, A[rt], false, Bh, (short)0, acc[rt][ct], false, false);
        acc[rt][ct] = __builtin_amdgcn_wmma_f32_16x16x32_bf16(
            false, A[rt], false, Bl, (short)0, acc[rt][ct], false, false);
      }
    }
  }

  // rowsum: lanes L and L+16 both hold partials of row lm
#pragma unroll
  for (int rt = 0; rt < 2; ++rt) {
    int tot = crow[rt] + __shfl_xor(crow[rt], 16, 32);
    if (lane < 16) cbuf[sp * N_TOK + r0 + rt * 16 + lm] = tot;
  }
  // P writeback: C tile vgpr g -> row g + 8*lh, col lm
  float* Pb = Pbuf + (long)sp * N_TOK * DDIM;
#pragma unroll
  for (int rt = 0; rt < 2; ++rt)
#pragma unroll
    for (int ct = 0; ct < 16; ++ct)
#pragma unroll
      for (int g = 0; g < 8; ++g) {
        long row = r0 + rt * 16 + g + 8 * lh;
        Pb[row * DDIM + ct * 16 + lm] = acc[rt][ct][g];
      }
}

// ---------------------------------------------------------------------------
// K6: two-weight softmax epilogue.  block = row, thread = col
// ---------------------------------------------------------------------------
__global__ void k6_epilogue(const float* __restrict__ Pbuf,
                            const int*   __restrict__ cbuf,
                            const float* __restrict__ s,
                            const float* __restrict__ colsum,
                            float* __restrict__ out) {
  const int row = blockIdx.x;
  const int d   = threadIdx.x;
  const float si = s[row];
  int c = 0;
#pragma unroll
  for (int sp = 0; sp < NSPLIT; ++sp) c += cbuf[sp * N_TOK + row];
  const float mx = (c > 0) ? fmaxf(si, NEG_FILL) : NEG_FILL;
  const float e1 = (c > 0) ? expf(si - mx) : 0.f;
  const float e0 = expf(NEG_FILL - mx);
  const float denom = (float)c * e1 + (float)(N_TOK - c) * e0;
  const float w1 = e1 / denom;
  const float w0 = e0 / denom;
  const long idx = (long)row * DDIM + d;
  float P = 0.f;
#pragma unroll
  for (int sp = 0; sp < NSPLIT; ++sp)
    P += Pbuf[(long)sp * N_TOK * DDIM + idx];
  out[idx] = (w1 - w0) * P + w0 * colsum[d];
}

// ---------------------------------------------------------------------------
extern "C" void kernel_launch(void* const* d_in, const int* in_sizes, int n_in,
                              void* d_out, int out_size, void* d_ws, size_t ws_size,
                              hipStream_t stream) {
  (void)in_sizes; (void)n_in; (void)out_size; (void)ws_size;
  const float* query = (const float*)d_in[0];
  const int*   mask  = (const int*)d_in[1];
  const float* Wq    = (const float*)d_in[2];
  const float* Wk    = (const float*)d_in[3];
  const float* Wv    = (const float*)d_in[4];
  const float* a     = (const float*)d_in[5];
  float* out = (float*)d_out;

  char* ws = (char*)d_ws;
  size_t off = 0;
  __bf16* vt_hi = (__bf16*)(ws + off); off += (size_t)DDIM * N_TOK * 2;
  __bf16* vt_lo = (__bf16*)(ws + off); off += (size_t)DDIM * N_TOK * 2;
  float*  Pbuf  = (float*)(ws + off);  off += (size_t)NSPLIT * N_TOK * DDIM * 4;
  int*    cbuf  = (int*)(ws + off);    off += (size_t)NSPLIT * N_TOK * 4;
  float*  sbuf  = (float*)(ws + off);  off += (size_t)N_TOK * 4;
  float*  csum  = (float*)(ws + off);  off += (size_t)DDIM * 4;
  float*  tbuf  = (float*)(ws + off);  off += (size_t)DDIM * 4;

  k1_fold_a<<<1, 256, 0, stream>>>(a, Wq, Wk, tbuf);
  k2_scores<<<N_TOK / 8, 256, 0, stream>>>(query, tbuf, sbuf);
  k3_vt<<<128, 256, 0, stream>>>(query, Wv, vt_hi, vt_lo);
  k4_colsum<<<DDIM, 256, 0, stream>>>(vt_hi, vt_lo, csum);
  k5_maskgemm<<<128, 256, 0, stream>>>(mask, vt_hi, vt_lo, Pbuf, cbuf);
  k6_epilogue<<<N_TOK, 256, 0, stream>>>(Pbuf, cbuf, sbuf, csum, out);
}
